// GRU_35570919145586
// MI455X (gfx1250) — matrix operand order repaired and checked by reading, not compiled, not run on hardware
//
#include <hip/hip_runtime.h>
#include <hip/hip_bf16.h>
#include <math.h>

// Problem sizes (match reference)
#define B_ 128
#define T_ 256
#define D_ 512
#define H_ 1024
#define C_ 512
#define N_ (B_ * T_)   // 32768 samples, n = b*T + t

typedef __bf16 bf16;
typedef __attribute__((ext_vector_type(16))) __bf16 v16bf;
typedef __attribute__((ext_vector_type(8)))  __bf16 v8bf;
typedef __attribute__((ext_vector_type(8)))  float  v8f;
typedef __attribute__((ext_vector_type(4)))  unsigned int v4u;
typedef __attribute__((ext_vector_type(8)))  int v8i;
typedef __attribute__((ext_vector_type(4)))  int v4i;

// ---------------------------------------------------------------------------
// WMMA fragment helpers (CDNA5 wave32 layouts, 16x16x32 bf16 -> f32)
//
// A (16x32, MxK) row-major source, base pre-offset to tile row 0:
//   lane m = lane&15 ; contiguous K-chunk [k+16*(lane>>4) .. +15]
// B (32x16, KxN) with source stored transposed [N][K] row-major:
//   lane n = lane&15 ; contiguous K-chunk [k+16*(lane>>4) .. +15]
// C/D (16x16 f32): vgpr r, lane L -> row r + 8*(L>>4), col L&15
// ---------------------------------------------------------------------------

__device__ __forceinline__ v16bf load_a16x32(const bf16* base, int ld, int k) {
  const int lane = threadIdx.x & 31;
  const int m  = lane & 15;
  const int ko = k + ((lane >> 4) << 4);
  const bf16* p = base + (size_t)m * ld + ko;
  v8bf lo = *(const v8bf*)p;          // K = ko .. ko+7
  v8bf hi = *(const v8bf*)(p + 8);    // K = ko+8 .. ko+15
  v16bf r;
#pragma unroll
  for (int i = 0; i < 8; ++i) { r[i] = lo[i]; r[i + 8] = hi[i]; }
  return r;
}

__device__ __forceinline__ v16bf load_b32x16(const bf16* __restrict__ base, int ld, int k) {
  const int lane = threadIdx.x & 31;
  const int n  = lane & 15;
  const int ko = k + ((lane >> 4) << 4);
  return *(const v16bf*)(base + (size_t)n * ld + ko);  // 32B-aligned chunk
}

__device__ __forceinline__ v8f wmma_bf16(v16bf a, v16bf b, v8f c) {
  return __builtin_amdgcn_wmma_f32_16x16x32_bf16(
      /*neg_a=*/false, a, /*neg_b=*/false, b,
      /*c_mod=*/(short)0, c, /*reuse_a=*/false, /*reuse_b=*/false);
}

// LDS fragment cache addressing: frag (ks, lane) stored as 16 bf16 (32 B)
__device__ __forceinline__ v16bf lds_frag(const bf16* sbase, int ks) {
  const int lane = threadIdx.x & 31;
  return *(const v16bf*)(sbase + ((size_t)ks * 32 + lane) * 16);
}
__device__ __forceinline__ void lds_frag_store(bf16* sbase, int ks, v16bf f) {
  const int lane = threadIdx.x & 31;
  *(v16bf*)(sbase + ((size_t)ks * 32 + lane) * 16) = f;
}

// ---------------------------------------------------------------------------
// Tensor Data Mover: DMA a 2D bf16 tile (rows x width) from global memory
// (row stride row_stride_elems) into LDS at byte offset lds_off.
// D# packing per CDNA5 ISA:
//   group0: [1:0]=count, [63:32]=lds_addr, [120:64]=global_addr, [127:126]=2
//   group1: [17:16]=data_size(1->2B), [79:48]=tensor_dim0, [111:80]=tensor_dim1,
//           [127:112]=tile_dim0, [143:128]=tile_dim1, [207:160]=dim0_stride
// Arity hedge: the gfx1250 TDM header ships only with the 6-arg toolchain.
// ---------------------------------------------------------------------------
__device__ __forceinline__ void tdm_load_2d(unsigned lds_off, const void* gaddr,
                                            unsigned width_elems, unsigned rows,
                                            unsigned row_stride_elems,
                                            unsigned tensor_rows) {
  const unsigned long long ga = (unsigned long long)gaddr;
  v4u g0;
  g0[0] = 1u;                                            // count=1, user mode
  g0[1] = lds_off;                                       // lds_addr (bytes)
  g0[2] = (unsigned)(ga & 0xFFFFFFFFu);                  // global_addr[31:0]
  g0[3] = (unsigned)((ga >> 32) & 0x01FFFFFFu) | (2u << 30);  // addr[56:32] | type=2
  v8i g1 = {};
  g1[0] = (int)(1u << 16);                               // data_size=1 (2 bytes)
  g1[1] = (int)((width_elems & 0xFFFFu) << 16);          // tensor_dim0[15:0]
  g1[2] = (int)(((width_elems >> 16) & 0xFFFFu) |        // tensor_dim0[31:16]
                ((tensor_rows & 0xFFFFu) << 16));        // tensor_dim1[15:0]
  g1[3] = (int)(((tensor_rows >> 16) & 0xFFFFu) |        // tensor_dim1[31:16]
                ((width_elems & 0xFFFFu) << 16));        // tile_dim0
  g1[4] = (int)(rows & 0xFFFFu);                         // tile_dim1 (tile_dim2=0)
  g1[5] = (int)row_stride_elems;                         // tensor_dim0_stride[31:0]
  g1[6] = 0;
  g1[7] = 0;
  v4i z4 = {};
#if __has_include(<hip/amd_detail/amd_gfx1250_TDM.h>)
  v8i z8 = {};
  __builtin_amdgcn_tensor_load_to_lds(g0, g1, z4, z4, z8, 0);
#else
  __builtin_amdgcn_tensor_load_to_lds(g0, g1, z4, z4, 0);
#endif
}

// ---------------------------------------------------------------------------
// Small utility kernels
// ---------------------------------------------------------------------------

__global__ void k_f2bf(const float* __restrict__ src, bf16* __restrict__ dst, int n) {
  int i = blockIdx.x * blockDim.x + threadIdx.x;
  if (i < n) dst[i] = (bf16)src[i];
}

// Embedding gather: xemb[n][d] = embed[x[n]][d]  (bf16), n = b*T + t
__global__ void k_embed_gather(const int* __restrict__ x, const float* __restrict__ embed,
                               bf16* __restrict__ xemb) {
  const int n   = blockIdx.x;
  const int tok = x[n];
  const float* src = embed + (size_t)tok * D_;
  bf16* dst = xemb + (size_t)n * D_;
  for (int d = threadIdx.x; d < D_; d += blockDim.x) dst[d] = (bf16)src[d];
}

// Zero initial hidden state (h^T layout [B][H]) and barrier counters
__global__ void k_init(bf16* __restrict__ h0, unsigned* __restrict__ bar) {
  int i = blockIdx.x * blockDim.x + threadIdx.x;
  if (i < B_ * H_) h0[i] = (bf16)0.0f;
  if (i < 8)       bar[i] = 0u;
}

// ---------------------------------------------------------------------------
// Input projections: wx[g][t][b][h] = sum_d Wg[g*H+h][d] * xemb[b*T+t][d]
// M = 3H = 3072 rows, N = 32768, K = 512. grid (192, 64), 256 threads.
// A tile staged per block into LDS fragments; B streamed with a register
// pipeline; sched_barrier keeps prefetched loads above the WMMA group.
// ---------------------------------------------------------------------------
__global__ void k_proj(const bf16* __restrict__ Wg,   // [3H][D] row-major (z,r,h)
                       const bf16* __restrict__ X,    // [N][D]
                       bf16* __restrict__ wx)         // [3][T][B][H]
{
  __shared__ bf16 sA[16 * 32 * 16];                 // 16 ks x 32 lanes x 32B = 16 KB
  const int wave = threadIdx.x >> 5;
  const int lane = threadIdx.x & 31;
  const int m0   = blockIdx.x * 16;                 // 0..3071
  const int n0   = blockIdx.y * 512 + wave * 64;    // 0..32767
  const bf16* Abase = Wg + (size_t)m0 * D_;

  for (int ks = wave; ks < 16; ks += 8)
    lds_frag_store(sA, ks, load_a16x32(Abase, D_, ks * 32));
  __syncthreads();

  const bf16* Xb[4] = { X + (size_t)(n0 +  0) * D_, X + (size_t)(n0 + 16) * D_,
                        X + (size_t)(n0 + 32) * D_, X + (size_t)(n0 + 48) * D_ };

  v8f acc[4] = {};
  v16bf bc[4], bn[4];
#pragma unroll
  for (int i = 0; i < 4; ++i) bc[i] = load_b32x16(Xb[i], D_, 0);

#pragma unroll 4
  for (int ks = 0; ks < 16; ++ks) {
    const int kn = (ks < 15) ? (ks + 1) * 32 : 0;   // harmless wrap prefetch
#pragma unroll
    for (int i = 0; i < 4; ++i) bn[i] = load_b32x16(Xb[i], D_, kn);
    v16bf a = lds_frag(sA, ks);
    __builtin_amdgcn_sched_barrier(0);   // keep next-iter loads above the WMMAs
    acc[0] = wmma_bf16(a, bc[0], acc[0]);
    acc[1] = wmma_bf16(a, bc[1], acc[1]);
    acc[2] = wmma_bf16(a, bc[2], acc[2]);
    acc[3] = wmma_bf16(a, bc[3], acc[3]);
#pragma unroll
    for (int i = 0; i < 4; ++i) bc[i] = bn[i];
  }

  // Store: per lane one 16B vector of 8 contiguous h values per tile
  const int nl = lane & 15;
  const int mo = (lane >> 4) << 3;
  const int g  = m0 >> 10;             // gate id (tile never straddles gates)
  const int hb = (m0 & (H_ - 1)) + mo; // h base, multiple of 8
#pragma unroll
  for (int i = 0; i < 4; ++i) {
    const int n = n0 + i * 16 + nl;
    const int b = n >> 8;        // n / T
    const int t = n & (T_ - 1);  // n % T
    v8bf ov;
#pragma unroll
    for (int r = 0; r < 8; ++r) ov[r] = (bf16)acc[i][r];
    *(v8bf*)(wx + (((size_t)g * T_ + t) * B_ + b) * H_ + hb) = ov;
  }
}

// ---------------------------------------------------------------------------
// Persistent GRU recurrence. 64 blocks x 256 threads = 512 waves, each wave
// owns one 16(H) x 16(B) tile of the hidden state. The block's three U tiles
// (16 rows x 1024 K x 3 = 96 KB) are DMA'd ONCE into LDS by the Tensor Data
// Mover and reused for all 256 steps; only the 32B/lane hidden fragment
// streams from global per k-step. Device-wide monotonic barrier per timestep.
// ---------------------------------------------------------------------------
__global__ void k_gru(const bf16* __restrict__ Uz,   // [H][H]
                      const bf16* __restrict__ Ur,   // [H][H]
                      const bf16* __restrict__ Uh,   // [H][H]
                      const bf16* __restrict__ wx,   // [3][T][B][H]
                      bf16* __restrict__ hbuf0,      // [B][H]  (h^T), holds h_init
                      bf16* __restrict__ hbuf1,      // [B][H]
                      unsigned* __restrict__ bar)
{
  extern __shared__ char smem_raw[];
  bf16* sUz = (bf16*)smem_raw;                 // [16][1024] row-major, 32 KB
  bf16* sUr = sUz + (size_t)16 * H_;
  bf16* sUh = sUr + (size_t)16 * H_;

  const int wave = threadIdx.x >> 5;
  const int lane = threadIdx.x & 31;
  const int h0 = blockIdx.x * 16;   // H tile
  const int b0 = wave * 16;         // B tile
  const int nl = lane & 15;
  const int mo = (lane >> 4) << 3;
  const unsigned nb = gridDim.x;

  // TDM: stage the three 16x1024 U tiles into LDS (wave 0 issues, all wait)
  if (wave == 0) {
    tdm_load_2d(0u,     Uz + (size_t)h0 * H_, H_, 16u, H_, H_);
    tdm_load_2d(32768u, Ur + (size_t)h0 * H_, H_, 16u, H_, H_);
    tdm_load_2d(65536u, Uh + (size_t)h0 * H_, H_, 16u, H_, H_);
    __builtin_amdgcn_s_wait_tensorcnt(0);
  }
  __syncthreads();

  const int bcol = b0 + nl;           // batch column owned by this lane
  const int hb   = h0 + mo;           // h base (8 contiguous values)

  for (int t = 0; t < T_; ++t) {
    const bf16* hc = (t & 1) ? hbuf1 : hbuf0;
    bf16*       hn = (t & 1) ? hbuf0 : hbuf1;
    const bf16* Bbase = hc + (size_t)b0 * H_;

    // Warm next timestep's gate rows while the GEMMs run
    if (t + 1 < T_) {
      __builtin_prefetch(wx + (((size_t)0 * T_ + (t + 1)) * B_ + bcol) * H_ + hb, 0, 1);
      __builtin_prefetch(wx + (((size_t)1 * T_ + (t + 1)) * B_ + bcol) * H_ + hb, 0, 1);
      __builtin_prefetch(wx + (((size_t)2 * T_ + (t + 1)) * B_ + bcol) * H_ + hb, 0, 1);
    }

    v8f accZ = {}, accR = {}, accH = {};
    v16bf bcur = load_b32x16(Bbase, H_, 0);
#pragma unroll 4
    for (int ks = 0; ks < 32; ++ks) {
      const int kn = (ks < 31) ? (ks + 1) * 32 : 0;
      v16bf bnxt = load_b32x16(Bbase, H_, kn);
      v16bf az = load_a16x32(sUz, H_, ks * 32);
      v16bf ar = load_a16x32(sUr, H_, ks * 32);
      v16bf ah = load_a16x32(sUh, H_, ks * 32);
      __builtin_amdgcn_sched_barrier(0);   // loads above, WMMAs below
      accZ = wmma_bf16(az, bcur, accZ);
      accR = wmma_bf16(ar, bcur, accR);
      accH = wmma_bf16(ah, bcur, accH);
      bcur = bnxt;
    }

    // Fused gate math: all per-lane vectors are 16B contiguous
    const v8bf wz = *(const v8bf*)(wx + (((size_t)0 * T_ + t) * B_ + bcol) * H_ + hb);
    const v8bf wr = *(const v8bf*)(wx + (((size_t)1 * T_ + t) * B_ + bcol) * H_ + hb);
    const v8bf wh = *(const v8bf*)(wx + (((size_t)2 * T_ + t) * B_ + bcol) * H_ + hb);
    const v8bf ho = *(const v8bf*)(hc + (size_t)bcol * H_ + hb);
    v8bf hv;
#pragma unroll
    for (int r = 0; r < 8; ++r) {
      const float z  = 1.0f / (1.0f + __expf(-(accZ[r] + (float)wz[r])));
      const float rr = 1.0f / (1.0f + __expf(-(accR[r] + (float)wr[r])));
      const float hh = tanhf((float)wh[r] + rr * accH[r]);
      hv[r] = (bf16)tanhf(z * (float)ho[r] + (1.0f - z) * hh);
    }
    *(v8bf*)(hn + (size_t)bcol * H_ + hb) = hv;

    // Device-wide barrier: monotonic counter, wait for nb*(t+1) arrivals.
    __threadfence();
    __syncthreads();
    if (threadIdx.x == 0) {
      __hip_atomic_fetch_add(&bar[0], 1u, __ATOMIC_ACQ_REL, __HIP_MEMORY_SCOPE_AGENT);
      const unsigned target = nb * (unsigned)(t + 1);
      while (__hip_atomic_load(&bar[0], __ATOMIC_ACQUIRE, __HIP_MEMORY_SCOPE_AGENT) < target) {
        __builtin_amdgcn_s_sleep(2);
      }
    }
    __syncthreads();
    __threadfence();
  }
}

// ---------------------------------------------------------------------------
// Final projection: logits[b][c] = sum_k Wph[c][k] * hT[b][k] + b_p[c]
// M = C = 512, N = B = 128, K = H = 1024. grid 32, 8 waves/block.
// ---------------------------------------------------------------------------
__global__ void k_logits(const bf16* __restrict__ Wph,  // [C][H]
                         const bf16* __restrict__ hT,   // [B][H] (final hidden)
                         const float* __restrict__ bp,  // [C]
                         float* __restrict__ logits)    // [B][C]
{
  __shared__ bf16 sA[32 * 32 * 16];                // 32 ks x 32B/lane = 32 KB
  const int wave = threadIdx.x >> 5;
  const int lane = threadIdx.x & 31;
  const int m0 = blockIdx.x * 16;  // class tile
  const int n0 = wave * 16;        // batch tile
  const bf16* Abase = Wph + (size_t)m0 * H_;
  const bf16* Bbase = hT + (size_t)n0 * H_;

  for (int ks = wave; ks < 32; ks += 8)
    lds_frag_store(sA, ks, load_a16x32(Abase, H_, ks * 32));
  __syncthreads();

  v8f acc = {};
  v16bf bcur = load_b32x16(Bbase, H_, 0);
#pragma unroll 4
  for (int ks = 0; ks < 32; ++ks) {
    const int kn = (ks < 31) ? (ks + 1) * 32 : 0;
    v16bf bnxt = load_b32x16(Bbase, H_, kn);
    v16bf a = lds_frag(sA, ks);
    __builtin_amdgcn_sched_barrier(0);
    acc = wmma_bf16(a, bcur, acc);
    bcur = bnxt;
  }

  const int nl = lane & 15;
  const int mo = (lane >> 4) << 3;
#pragma unroll
  for (int r = 0; r < 8; ++r) {
    const int m = m0 + r + mo;
    const int n = n0 + nl;
    logits[(size_t)n * C_ + m] = acc[r] + bp[m];
  }
}

// log_softmax over C per batch row; one block per b.
__global__ void k_logsoftmax(const float* __restrict__ logits, float* __restrict__ out) {
  __shared__ float red[256];
  const int b = blockIdx.x;
  const float* row = logits + (size_t)b * C_;

  float m = -1e30f;
  for (int c = threadIdx.x; c < C_; c += blockDim.x) m = fmaxf(m, row[c]);
  red[threadIdx.x] = m;
  __syncthreads();
  for (int s = 128; s > 0; s >>= 1) {
    if (threadIdx.x < (unsigned)s) red[threadIdx.x] = fmaxf(red[threadIdx.x], red[threadIdx.x + s]);
    __syncthreads();
  }
  m = red[0];
  __syncthreads();

  float sum = 0.0f;
  for (int c = threadIdx.x; c < C_; c += blockDim.x) sum += __expf(row[c] - m);
  red[threadIdx.x] = sum;
  __syncthreads();
  for (int s = 128; s > 0; s >>= 1) {
    if (threadIdx.x < (unsigned)s) red[threadIdx.x] += red[threadIdx.x + s];
    __syncthreads();
  }
  const float lse = m + __logf(red[0]);
  for (int c = threadIdx.x; c < C_; c += blockDim.x) out[(size_t)b * C_ + c] = row[c] - lse;
}

// ---------------------------------------------------------------------------
// Host launcher
// ---------------------------------------------------------------------------
extern "C" void kernel_launch(void* const* d_in, const int* in_sizes, int n_in,
                              void* d_out, int out_size, void* d_ws, size_t ws_size,
                              hipStream_t stream) {
  (void)in_sizes; (void)n_in; (void)out_size; (void)ws_size;

  const int*   x     = (const int*)d_in[0];    // [B,T]
  const float* embed = (const float*)d_in[1];  // [D,D]
  const float* W_z   = (const float*)d_in[2];  // [H,D]
  const float* U_z   = (const float*)d_in[3];  // [H,H]
  const float* W_r   = (const float*)d_in[4];  // [H,D]
  const float* U_r   = (const float*)d_in[5];  // [H,H]
  const float* W_m   = (const float*)d_in[6];  // [H,D]
  const float* U_m   = (const float*)d_in[7];  // [H,H]
  const float* W_ph  = (const float*)d_in[8];  // [C,H]
  const float* b_p   = (const float*)d_in[9];  // [C,1]
  float* out = (float*)d_out;                  // [B,C]

  char* p = (char*)d_ws;
  auto take = [&](size_t bytes) -> void* {
    void* r = (void*)p;
    p += (bytes + 255) & ~(size_t)255;
    return r;
  };

  bf16* xemb   = (bf16*)take((size_t)N_ * D_ * 2);          // [N][D]
  bf16* Wg     = (bf16*)take((size_t)3 * H_ * D_ * 2);      // [3H][D]
  bf16* Ug     = (bf16*)take((size_t)3 * H_ * H_ * 2);      // [3H][H]
  bf16* Wphbf  = (bf16*)take((size_t)C_ * H_ * 2);          // [C][H]
  bf16* wx     = (bf16*)take((size_t)3 * T_ * B_ * H_ * 2); // [3][T][B][H]
  bf16* hbuf0  = (bf16*)take((size_t)B_ * H_ * 2);
  bf16* hbuf1  = (bf16*)take((size_t)B_ * H_ * 2);
  float* logits = (float*)take((size_t)B_ * C_ * 4);
  unsigned* bar = (unsigned*)take(256);

  const int HD = H_ * D_, HH = H_ * H_;
  // Weight conversions to bf16 (z, r, h order in the concatenated blocks)
  k_f2bf<<<(HD + 255) / 256, 256, 0, stream>>>(W_z, Wg + 0 * (size_t)HD, HD);
  k_f2bf<<<(HD + 255) / 256, 256, 0, stream>>>(W_r, Wg + 1 * (size_t)HD, HD);
  k_f2bf<<<(HD + 255) / 256, 256, 0, stream>>>(W_m, Wg + 2 * (size_t)HD, HD);
  k_f2bf<<<(HH + 255) / 256, 256, 0, stream>>>(U_z, Ug + 0 * (size_t)HH, HH);
  k_f2bf<<<(HH + 255) / 256, 256, 0, stream>>>(U_r, Ug + 1 * (size_t)HH, HH);
  k_f2bf<<<(HH + 255) / 256, 256, 0, stream>>>(U_m, Ug + 2 * (size_t)HH, HH);
  k_f2bf<<<(C_ * H_ + 255) / 256, 256, 0, stream>>>(W_ph, Wphbf, C_ * H_);

  // Embedding gather + hidden/barrier init
  k_embed_gather<<<N_, 256, 0, stream>>>(x, embed, xemb);
  k_init<<<(B_ * H_ + 255) / 256, 256, 0, stream>>>(hbuf0, bar);

  // Input-side projections for all timesteps
  k_proj<<<dim3(192, 64), 256, 0, stream>>>(Wg, xemb, wx);

  // Persistent recurrence (96 KB dynamic LDS, TDM-staged U tiles)
  k_gru<<<64, 256, 3 * 16 * H_ * 2, stream>>>(Ug + 0 * (size_t)HH, Ug + 1 * (size_t)HH,
                                              Ug + 2 * (size_t)HH, wx, hbuf0, hbuf1, bar);

  // Head: logits (T_=256 even -> final hidden state lives in hbuf0)
  k_logits<<<32, 256, 0, stream>>>(Wphbf, hbuf0, b_p, logits);
  k_logsoftmax<<<128, 256, 0, stream>>>(logits, out);
}
